// CrossAttLayer_87729001988843
// MI455X (gfx1250) — compile-verified, matrix operand order checked by reading, not run on hardware
//
#include <hip/hip_runtime.h>
#include <hip/hip_bf16.h>
#include <math.h>
#include <stdint.h>

typedef __attribute__((ext_vector_type(16))) __bf16 v16bf;
typedef __attribute__((ext_vector_type(8)))  __bf16 v8bf;
typedef __attribute__((ext_vector_type(8)))  float  v8f;
typedef int v4i_g __attribute__((vector_size(16)));  // matches async-LDS builtin param

#define T_    91
#define D_    256
#define H_    4
#define DH_   64
#define NA_   128
#define NRG_  1024
#define KEXP_ 4
#define NEGF  (-1.0e10f)

#define AS1 __attribute__((address_space(1)))
#define AS3 __attribute__((address_space(3)))

#if defined(__HIP_DEVICE_COMPILE__) && __has_builtin(__builtin_amdgcn_global_load_async_to_lds_b128)
#define ASYNC_LDS 1
#else
#define ASYNC_LDS 0
#endif

enum { EPI_SCATTER = 0, EPI_S = 1, EPI_RELU_BF16 = 2, EPI_RELU_F32 = 3 };

__device__ __forceinline__ v8f vzero8() {
  v8f z = {0.f, 0.f, 0.f, 0.f, 0.f, 0.f, 0.f, 0.f};
  return z;
}

// 16-byte global->LDS copy: async DMA (ASYNCcnt) when available, else via VGPRs.
__device__ __forceinline__ void cp16_g2l(const __bf16* src, __bf16* dst_lds) {
#if ASYNC_LDS
  __builtin_amdgcn_global_load_async_to_lds_b128(
      (AS1 v4i_g*)(uintptr_t)src,
      (AS3 v4i_g*)(uintptr_t)(unsigned)(uintptr_t)dst_lds, 0, 0);
#else
  *(v8bf*)dst_lds = *(const v8bf*)src;
#endif
}

__device__ __forceinline__ void wait_async(int pending) {
#if ASYNC_LDS
#if __has_builtin(__builtin_amdgcn_s_wait_asynccnt)
  if (pending) __builtin_amdgcn_s_wait_asynccnt(2);
  else         __builtin_amdgcn_s_wait_asynccnt(0);
#else
  if (pending) asm volatile("s_wait_asynccnt 0x2" ::: "memory");
  else         asm volatile("s_wait_asynccnt 0x0" ::: "memory");
#endif
#else
  (void)pending;
#endif
}

// Load one 16x32 bf16 A/B^T fragment row for this lane.
// Per ISA 7.12.2 (16-bit A 16x32): lane holds row m=lane&15; element e maps to
// k = kstep + (e/8)*16 + (lane>>4)*8 + (e%8)  -> two contiguous 8-elem (16B) loads.
__device__ __forceinline__ v16bf load_frag_row(const __bf16* rowbase, int kstep, int half) {
  const __bf16* p = rowbase + kstep + half * 8;
  v8bf lo = *(const v8bf*)(p);
  v8bf hi = *(const v8bf*)(p + 16);
  v16bf r;
#pragma unroll
  for (int i = 0; i < 8; ++i) { r[i] = lo[i]; r[i + 8] = hi[i]; }
  return r;
}

// ---------------- LayerNorm: one wave per row of 256 ----------------
__global__ __launch_bounds__(256) void ln_kernel(const float* __restrict__ x,
                                                 const float* __restrict__ g,
                                                 const float* __restrict__ b,
                                                 float* __restrict__ outF,
                                                 __bf16* __restrict__ outB, int rows) {
  int row  = blockIdx.x * 8 + (threadIdx.x >> 5);
  int lane = threadIdx.x & 31;
  if (row >= rows) return;
  const float* xr = x + (size_t)row * D_;
  float v[8];
  float s = 0.f;
#pragma unroll
  for (int i = 0; i < 8; ++i) { v[i] = xr[lane + i * 32]; s += v[i]; }
#pragma unroll
  for (int m = 1; m < 32; m <<= 1) s += __shfl_xor(s, m, 32);
  float mean = s * (1.0f / D_);
  float s2 = 0.f;
#pragma unroll
  for (int i = 0; i < 8; ++i) { float d = v[i] - mean; s2 += d * d; }
#pragma unroll
  for (int m = 1; m < 32; m <<= 1) s2 += __shfl_xor(s2, m, 32);
  float rstd = rsqrtf(s2 * (1.0f / D_) + 1e-5f);
#pragma unroll
  for (int i = 0; i < 8; ++i) {
    int c = lane + i * 32;
    float y = (v[i] - mean) * rstd * g[c] + b[c];
    if (outF) outF[(size_t)row * D_ + c] = y;
    if (outB) outB[(size_t)row * D_ + c] = (__bf16)y;
  }
}

// ------------- weight convert+transpose: w[K][N] f32 -> wt[N][K] bf16 -------------
__global__ __launch_bounds__(256) void wtrans_kernel(const float* __restrict__ w,
                                                     __bf16* __restrict__ wt, int K, int N) {
  int idx = blockIdx.x * 256 + threadIdx.x;
  if (idx >= K * N) return;
  int n = idx / K, k = idx - n * K;
  wt[idx] = (__bf16)w[(size_t)k * N + n];
}

// ---------------- generic WMMA bf16 GEMM, one wave per 16x64 tile ----------------
// A: [M][Kd] bf16 row-major.  Bt: [N][Kd] bf16 (weights pre-transposed).
// Per k-step: issue all 10 b128 loads (A + 4xB) up front, then 4 back-to-back
// WMMAs reusing the A fragment -> compiler can pipeline waits across steps.
__global__ __launch_bounds__(32) void gemm_wmma_kernel(
    const __bf16* __restrict__ A, const __bf16* __restrict__ Bt,
    const float* __restrict__ bias, int N, int Kd, int epi,
    float* __restrict__ outF, __bf16* __restrict__ outB,
    const float* __restrict__ qscale, const float* __restrict__ resid,
    __bf16* __restrict__ auxB, int aux_trans, int rsub) {
  int lane = threadIdx.x;
  int half = lane >> 4, l15 = lane & 15;
  int m0 = blockIdx.x * 16, n0 = blockIdx.y * 64;
  const __bf16* arow = A + (size_t)(m0 + l15) * Kd;
  const __bf16* brow0 = Bt + (size_t)(n0 + 0 * 16 + l15) * Kd;
  const __bf16* brow1 = Bt + (size_t)(n0 + 1 * 16 + l15) * Kd;
  const __bf16* brow2 = Bt + (size_t)(n0 + 2 * 16 + l15) * Kd;
  const __bf16* brow3 = Bt + (size_t)(n0 + 3 * 16 + l15) * Kd;
  v8f acc[4];
#pragma unroll
  for (int j = 0; j < 4; ++j) acc[j] = vzero8();
  for (int k = 0; k < Kd; k += 32) {
    v16bf af  = load_frag_row(arow,  k, half);
    v16bf bf0 = load_frag_row(brow0, k, half);
    v16bf bf1 = load_frag_row(brow1, k, half);
    v16bf bf2 = load_frag_row(brow2, k, half);
    v16bf bf3 = load_frag_row(brow3, k, half);
    acc[0] = __builtin_amdgcn_wmma_f32_16x16x32_bf16(false, af, false, bf0, (short)0,
                                                     acc[0], false, false);
    acc[1] = __builtin_amdgcn_wmma_f32_16x16x32_bf16(false, af, false, bf1, (short)0,
                                                     acc[1], false, false);
    acc[2] = __builtin_amdgcn_wmma_f32_16x16x32_bf16(false, af, false, bf2, (short)0,
                                                     acc[2], false, false);
    acc[3] = __builtin_amdgcn_wmma_f32_16x16x32_bf16(false, af, false, bf3, (short)0,
                                                     acc[3], false, false);
  }
#pragma unroll
  for (int j = 0; j < 4; ++j) {
    int n = n0 + j * 16 + l15;
    float bn = bias[n];
#pragma unroll
    for (int r = 0; r < 8; ++r) {
      int m = m0 + r + 8 * half;  // C-frag: row m = r + 8*half, col n = lane&15
      float vv = fmaxf(acc[j][r] + bn, 0.f);
      if (epi == EPI_SCATTER) {  // rows are (src,t) pairs; scatter to (t,h,src,dh)
        if (qscale) vv *= qscale[n & (DH_ - 1)];
        int rr = m / T_, t = m - rr * T_;
        int h = n >> 6, dh = n & 63;
        size_t th = (size_t)t * H_ + h;
        outF[(th * rsub + rr) * DH_ + dh] = vv;
        if (auxB) {
          if (aux_trans) auxB[(th * DH_ + dh) * rsub + rr] = (__bf16)vv;  // [th][dh][r]
          else           auxB[(th * rsub + rr) * DH_ + dh] = (__bf16)vv;  // [th][r][dh]
        }
      } else if (epi == EPI_S) {
        size_t o = (size_t)m * N + n;
        outB[o] = (__bf16)(vv + resid[o]);
      } else if (epi == EPI_RELU_BF16) {
        outB[(size_t)m * N + n] = (__bf16)vv;
      } else {
        outF[(size_t)m * N + n] = vv;
      }
    }
  }
}

// ---------------- flash attention per (t,h): 128 Q rows x 1024 K/V rows ----------------
// K/V/Q staged from bf16 side copies via (double-buffered) async global->LDS DMA.
__global__ __launch_bounds__(256) void attn_kernel(
    const __bf16* __restrict__ Qb,  // [th][a][dh]
    const __bf16* __restrict__ Kb,  // [th][r][dh]
    const __bf16* __restrict__ Vt,  // [th][dh][r]
    const unsigned char* __restrict__ arm, const unsigned char* __restrict__ pm,
    const unsigned char* __restrict__ rvm, __bf16* __restrict__ Y1b) {
  int th = blockIdx.x;
  int t = th >> 2, h = th & 3;
  int tid = threadIdx.x;
  int w = tid >> 5, lane = tid & 31, half = lane >> 4, l15 = lane & 15;

  __shared__ __align__(16) __bf16 Qs[NA_ * DH_];      // 16 KB, [a][dh]
  __shared__ __align__(16) __bf16 Ks[2][32 * DH_];    //  8 KB, [r_local][dh]
  __shared__ __align__(16) __bf16 VsT[2][DH_ * 32];   //  8 KB, [dh][r_local]
  __shared__ __align__(16) __bf16 Ps[8][16 * 32];     //  8 KB, per-wave P tile

  const __bf16* Qg = Qb + (size_t)th * NA_ * DH_;
  const __bf16* Kg = Kb + (size_t)th * NRG_ * DH_;
  const __bf16* Vg = Vt + (size_t)th * DH_ * NRG_;

  // Qs: 1024 x 16B chunks, 4 per thread (async, completes before first wait(<=2))
#pragma unroll
  for (int j = 0; j < 4; ++j) {
    int c = tid + j * 256;
    cp16_g2l(Qg + c * 8, Qs + c * 8);
  }
  // stage one 32-row K/V chunk into LDS buffer `buf` (2 async ops per thread)
  auto issue_chunk = [&](int rc, int buf) {
    int r0 = rc * 32;
    cp16_g2l(Kg + (size_t)r0 * DH_ + tid * 8, &Ks[buf][tid * 8]);
    int dh = tid >> 2, part = tid & 3;
    cp16_g2l(Vg + (size_t)dh * NRG_ + r0 + part * 8, &VsT[buf][dh * 32 + part * 8]);
  };
  issue_chunk(0, 0);

  int a0 = w * 16;  // this wave's 16 Q rows
  float mrow[8], lrow[8];
  v8f O[4];
#pragma unroll
  for (int r = 0; r < 8; ++r) { mrow[r] = -INFINITY; lrow[r] = 0.f; }
#pragma unroll
  for (int f = 0; f < 4; ++f) O[f] = vzero8();

  const __bf16* qrow = Qs + (size_t)(a0 + l15) * DH_;
  const int NC = NRG_ / 32;

  for (int rc = 0; rc < NC; ++rc) {
    int buf = rc & 1;
    int prefetch = (rc + 1 < NC);
    if (prefetch) issue_chunk(rc + 1, buf ^ 1);  // writes buffer retired at rc-1
    wait_async(prefetch);                        // chunk rc resident (in-order)
    __syncthreads();                             // ... in every thread

    // energy tiles: E = Q(16x64) @ K^T(64x32) -> two 16x16 accum frags
    v8f e0 = vzero8(), e1 = vzero8();
#pragma unroll
    for (int kk = 0; kk < DH_; kk += 32) {
      v16bf af = load_frag_row(qrow, kk, half);
      v16bf b0 = load_frag_row(&Ks[buf][l15 * DH_], kk, half);
      v16bf b1 = load_frag_row(&Ks[buf][(16 + l15) * DH_], kk, half);
      e0 = __builtin_amdgcn_wmma_f32_16x16x32_bf16(false, af, false, b0, (short)0, e0,
                                                   false, false);
      e1 = __builtin_amdgcn_wmma_f32_16x16x32_bf16(false, af, false, b1, (short)0, e1,
                                                   false, false);
    }

    // scale + mask + online softmax update
    int r0 = rc * 32;
#pragma unroll
    for (int r = 0; r < 8; ++r) {
      int ai = a0 + r + 8 * half;
      int rg0 = r0 + l15, rg1 = rg0 + 16;
      float x0 = e0[r] * 0.5f, x1 = e1[r] * 0.5f;  // /SCALE, SCALE=2
      const unsigned char* armrow = arm + ((size_t)t * NA_ + ai) * NRG_;
      bool va = rvm[t * NA_ + ai] != 0;
      if (!(va && armrow[rg0] && pm[t * NRG_ + rg0])) x0 = NEGF;
      if (!(va && armrow[rg1] && pm[t * NRG_ + rg1])) x1 = NEGF;
      float mx = fmaxf(x0, x1);
#pragma unroll
      for (int s = 1; s < 16; s <<= 1) mx = fmaxf(mx, __shfl_xor(mx, s, 32));
      float mnew  = fmaxf(mrow[r], mx);
      float alpha = __expf(mrow[r] - mnew);
      float p0 = __expf(x0 - mnew), p1 = __expf(x1 - mnew);
      float ps = p0 + p1;
#pragma unroll
      for (int s = 1; s < 16; s <<= 1) ps += __shfl_xor(ps, s, 32);
      lrow[r] = lrow[r] * alpha + ps;
      mrow[r] = mnew;
#pragma unroll
      for (int f = 0; f < 4; ++f) O[f][r] *= alpha;
      int mloc = r + 8 * half;
      Ps[w][mloc * 32 + l15]      = (__bf16)p0;
      Ps[w][mloc * 32 + 16 + l15] = (__bf16)p1;
    }

    // O += P(16x32) @ V(32x64); per-wave LDS RAW is in-order, no barrier needed
    {
      v16bf aP = load_frag_row(&Ps[w][l15 * 32], 0, half);
#pragma unroll
      for (int f = 0; f < 4; ++f) {
        v16bf bV = load_frag_row(&VsT[buf][(f * 16 + l15) * 32], 0, half);
        O[f] = __builtin_amdgcn_wmma_f32_16x16x32_bf16(false, aP, false, bV, (short)0, O[f],
                                                       false, false);
      }
    }
    __syncthreads();  // all waves done with buf before it is overwritten
  }

  // normalize and write Y1 (bf16, [a][t][h*64+dh] row-major over D)
#pragma unroll
  for (int f = 0; f < 4; ++f) {
#pragma unroll
    for (int r = 0; r < 8; ++r) {
      int ai = a0 + r + 8 * half;
      int dh = f * 16 + l15;
      Y1b[((size_t)ai * T_ + t) * D_ + h * DH_ + dh] = (__bf16)(O[f][r] / lrow[r]);
    }
  }
}

// ------------------------------------------------------------------
extern "C" void kernel_launch(void* const* d_in, const int* in_sizes, int n_in,
                              void* d_out, int out_size, void* d_ws, size_t ws_size,
                              hipStream_t stream) {
  const float* agent   = (const float*)d_in[0];
  const float* rg      = (const float*)d_in[1];
  const float* ln_x_g  = (const float*)d_in[2];
  const float* ln_x_b  = (const float*)d_in[3];
  const float* wk      = (const float*)d_in[4];
  const float* bk      = (const float*)d_in[5];
  const float* wv      = (const float*)d_in[6];
  const float* bv      = (const float*)d_in[7];
  const float* wq      = (const float*)d_in[8];
  const float* bq      = (const float*)d_in[9];
  const float* q_scale = (const float*)d_in[10];
  const float* wy      = (const float*)d_in[11];
  const float* by      = (const float*)d_in[12];
  const float* wf1     = (const float*)d_in[13];
  const float* bf1     = (const float*)d_in[14];
  const float* wf2     = (const float*)d_in[15];
  const float* bf2     = (const float*)d_in[16];
  const float* ln_z_g  = (const float*)d_in[17];
  const float* ln_z_b  = (const float*)d_in[18];
  const unsigned char* arm = (const unsigned char*)d_in[19];
  const unsigned char* pm  = (const unsigned char*)d_in[20];
  const unsigned char* rvm = (const unsigned char*)d_in[21];

  const size_t ZN = (size_t)NA_ * T_ * D_;        // 2,981,888
  const size_t QN = (size_t)T_ * H_ * NA_ * DH_;  // 2,981,888
  const size_t KN = (size_t)T_ * H_ * NRG_ * DH_; // 23,855,104
  float* Z  = (float*)d_out;
  float* Qp = Z + ZN;
  float* Kp = Qp + QN;
  float* Vp = Kp + KN;

  char* wsp = (char*)d_ws;
  auto carve = [&](size_t bytes) -> void* {
    void* p = (void*)wsp;
    wsp += (bytes + 255) & ~(size_t)255;
    return p;
  };
  float*  agentnF = (float*)carve(ZN * 4);
  __bf16* agentnB = (__bf16*)carve(ZN * 2);
  __bf16* rgnB    = (__bf16*)carve((size_t)NRG_ * T_ * D_ * 2);
  __bf16* wkT     = (__bf16*)carve((size_t)D_ * D_ * 2);
  __bf16* wvT     = (__bf16*)carve((size_t)D_ * D_ * 2);
  __bf16* wqT     = (__bf16*)carve((size_t)D_ * D_ * 2);
  __bf16* wyT     = (__bf16*)carve((size_t)D_ * D_ * 2);
  __bf16* wf1T    = (__bf16*)carve((size_t)D_ * KEXP_ * D_ * 2);
  __bf16* wf2T    = (__bf16*)carve((size_t)KEXP_ * D_ * D_ * 2);
  __bf16* Qb      = (__bf16*)carve(QN * 2);  // bf16 Q, [th][a][dh]
  __bf16* Kb      = (__bf16*)carve(KN * 2);  // bf16 K, [th][r][dh]
  __bf16* Vt      = (__bf16*)carve(KN * 2);  // bf16 V, [th][dh][r]
  __bf16* Y1b     = (__bf16*)carve(ZN * 2);
  __bf16* Sb      = (__bf16*)carve(ZN * 2);
  __bf16* F1b     = (__bf16*)carve((size_t)NA_ * T_ * KEXP_ * D_ * 2);
  float*  F2f     = (float*)carve(ZN * 4);

  // weights -> bf16 [N][K]
  wtrans_kernel<<<(D_ * D_ + 255) / 256, 256, 0, stream>>>(wk, wkT, D_, D_);
  wtrans_kernel<<<(D_ * D_ + 255) / 256, 256, 0, stream>>>(wv, wvT, D_, D_);
  wtrans_kernel<<<(D_ * D_ + 255) / 256, 256, 0, stream>>>(wq, wqT, D_, D_);
  wtrans_kernel<<<(D_ * D_ + 255) / 256, 256, 0, stream>>>(wy, wyT, D_, D_);
  wtrans_kernel<<<(D_ * KEXP_ * D_ + 255) / 256, 256, 0, stream>>>(wf1, wf1T, D_, KEXP_ * D_);
  wtrans_kernel<<<(KEXP_ * D_ * D_ + 255) / 256, 256, 0, stream>>>(wf2, wf2T, KEXP_ * D_, D_);

  // layernorms
  ln_kernel<<<(NA_ * T_) / 8, 256, 0, stream>>>(agent, ln_x_g, ln_x_b, agentnF, agentnB,
                                                NA_ * T_);
  ln_kernel<<<(NRG_ * T_) / 8, 256, 0, stream>>>(rg, ln_x_g, ln_x_b, nullptr, rgnB,
                                                 NRG_ * T_);

  // K, V, Q projections (relu [+q_scale]); f32 into d_out (t,h,src,dh) + bf16 side copies
  dim3 gKV((NRG_ * T_) / 16, D_ / 64);
  gemm_wmma_kernel<<<gKV, 32, 0, stream>>>(rgnB, wkT, bk, D_, D_, EPI_SCATTER, Kp, nullptr,
                                           nullptr, nullptr, Kb, 0, NRG_);
  gemm_wmma_kernel<<<gKV, 32, 0, stream>>>(rgnB, wvT, bv, D_, D_, EPI_SCATTER, Vp, nullptr,
                                           nullptr, nullptr, Vt, 1, NRG_);
  dim3 gA((NA_ * T_) / 16, D_ / 64);
  gemm_wmma_kernel<<<gA, 32, 0, stream>>>(agentnB, wqT, bq, D_, D_, EPI_SCATTER, Qp, nullptr,
                                          q_scale, nullptr, Qb, 0, NA_);

  // fused masked-softmax attention -> Y1 (bf16)
  attn_kernel<<<T_ * H_, 256, 0, stream>>>(Qb, Kb, Vt, arm, pm, rvm, Y1b);

  // S = relu(Y1@wy+by) + agent_n
  gemm_wmma_kernel<<<gA, 32, 0, stream>>>(Y1b, wyT, by, D_, D_, EPI_S, nullptr, Sb, nullptr,
                                          agentnF, nullptr, 0, 0);
  // F1 = relu(S@wf1+bf1)
  dim3 gF1((NA_ * T_) / 16, (KEXP_ * D_) / 64);
  gemm_wmma_kernel<<<gF1, 32, 0, stream>>>(Sb, wf1T, bf1, KEXP_ * D_, D_, EPI_RELU_BF16,
                                           nullptr, F1b, nullptr, nullptr, nullptr, 0, 0);
  // F2 = relu(F1@wf2+bf2)
  gemm_wmma_kernel<<<gA, 32, 0, stream>>>(F1b, wf2T, bf2, D_, KEXP_ * D_, EPI_RELU_F32, F2f,
                                          nullptr, nullptr, nullptr, nullptr, 0, 0);
  // Z = LN(F2)
  ln_kernel<<<(NA_ * T_) / 8, 256, 0, stream>>>(F2f, ln_z_g, ln_z_b, Z, nullptr, NA_ * T_);

  (void)in_sizes; (void)n_in; (void)out_size; (void)ws_size;
}
